// AttentionHead_9371618640080
// MI455X (gfx1250) — matrix operand order repaired and checked by reading, not compiled, not run on hardware
//
#include <hip/hip_runtime.h>
#include <hip/hip_bf16.h>

typedef __bf16 bf16_t;
typedef __attribute__((ext_vector_type(16))) __bf16 v16bf;
typedef __attribute__((ext_vector_type(8)))  float  v8f;

#define N_EMB 1024
#define HS    64
#define TSEQ  4096
#define BATCH 8

__device__ __forceinline__ v8f wmma_bf16(v16bf a, v16bf b, v8f c) {
  // D = A(16x32 bf16) * B(32x16 bf16) + C(16x16 f32)
  return __builtin_amdgcn_wmma_f32_16x16x32_bf16(
      /*neg_a=*/false, a, /*neg_b=*/false, b,
      /*c_mod=*/(short)0, c, /*reuse_a=*/false, /*reuse_b=*/false);
}

// A-matrix (16x32, 16-bit) per-lane K base for vector element pair i (ISA 7.12.2):
// VGPR i<4: K=2i,2i+1 (+8 for lanes 16..31); VGPR i>=4: K=16+2(i-4),... (+8)
__device__ __forceinline__ int a_kbase(int i, int half4) {
  return 2 * (i & 3) + ((i & 4) << 2) + 8 * half4;
}

// ---------------- Kernel 1: fused Q/K/V projection (fp32 -> bf16 WMMA -> bf16) ---
// q,k stored row-major [B*T][64]; v stored TRANSPOSED [B][64][T] so the
// attention PV B-operand is a contiguous 32B global load (no LDS staging).
__global__ __launch_bounds__(256) void proj_kernel(
    const float* __restrict__ x,
    const float* __restrict__ Wq, const float* __restrict__ Wk,
    const float* __restrict__ Wv,
    bf16_t* __restrict__ qo, bf16_t* __restrict__ ko, bf16_t* __restrict__ vo)
{
  // W chunk staged transposed: Wt[m][h*32 + 16*((kk>>3)&1) + 8*(kk>>4) + (kk&7)]  (kk = k within 32-chunk)
  __shared__ __align__(32) bf16_t Wt[3][HS * 32];

  const int tid   = threadIdx.x;
  const int wave  = tid >> 5;
  const int lane  = tid & 31;
  const int half4 = lane >> 4;
  const int l15   = lane & 15;
  const int rowb  = blockIdx.x * 128 + wave * 16;   // 16 rows per wave

  const float* Ws[3] = {Wq, Wk, Wv};

  v8f zero = {};
  v8f acc[3][4];
  #pragma unroll
  for (int m = 0; m < 3; ++m)
    #pragma unroll
    for (int t = 0; t < 4; ++t) acc[m][t] = zero;

  for (int kc = 0; kc < N_EMB / 32; ++kc) {
    __syncthreads();
    // cooperative stage of 3 x (32 x 64) fp32 W chunk -> bf16 transposed LDS
    for (int idx = tid; idx < 3 * 2048; idx += 256) {
      int m  = idx >> 11;
      int e  = idx & 2047;
      int kk = e >> 6;
      int h  = e & 63;
      Wt[m][h * 32 + ((kk >> 3) & 1) * 16 + (kk >> 4) * 8 + (kk & 7)] = (bf16_t)Ws[m][(size_t)(kc * 32 + kk) * HS + h];
    }
    __syncthreads();

    // A tile: x rows, fp32 -> bf16, interleaved A layout
    v16bf a;
    const float* xrow = x + (size_t)(rowb + l15) * N_EMB + kc * 32;
    #pragma unroll
    for (int i = 0; i < 8; ++i) {
      int kk = a_kbase(i, half4);
      a[2 * i]     = (bf16_t)xrow[kk];
      a[2 * i + 1] = (bf16_t)xrow[kk + 1];
    }

    #pragma unroll
    for (int m = 0; m < 3; ++m) {
      #pragma unroll
      for (int t = 0; t < 4; ++t) {
        int h = t * 16 + l15;                 // B col (n) = lane&15
        const v16bf b = *(const v16bf*)&Wt[m][h * 32 + 16 * half4];
        acc[m][t] = wmma_bf16(a, b, acc[m][t]);
      }
    }
  }

  #pragma unroll
  for (int t = 0; t < 4; ++t)
    #pragma unroll
    for (int i = 0; i < 8; ++i) {
      int row = rowb + i + 8 * half4;         // C/D layout: M = vgpr + 8*half
      int col = t * 16 + l15;
      qo[(size_t)row * HS + col] = (bf16_t)acc[0][t][i];
      ko[(size_t)row * HS + col] = (bf16_t)acc[1][t][i];
      int bb = row >> 12;                     // row / TSEQ
      int tt = row & (TSEQ - 1);
      vo[((size_t)bb * HS + col) * TSEQ + tt] = (bf16_t)acc[2][t][i];
    }
}

// ---------------- Kernel 2: causal flash attention (barrier-free) -----------------
__global__ __launch_bounds__(128) void attn_kernel(
    const bf16_t* __restrict__ q, const bf16_t* __restrict__ k,
    const bf16_t* __restrict__ vt, float* __restrict__ out)
{
  __shared__ __align__(32) bf16_t Pl[4][16 * 64];  // per-wave P scratch [row][key]

  const int tid   = threadIdx.x;
  const int wave  = tid >> 5;
  const int lane  = tid & 31;
  const int half4 = lane >> 4;
  const int l15   = lane & 15;

  const int b  = blockIdx.x >> 6;            // T/64 = 64 query tiles per batch
  const int qt = blockIdx.x & 63;
  const int qw = qt * 64 + wave * 16;        // this wave's query base

  const size_t base = (size_t)b * TSEQ * HS;
  const bf16_t* qp = q + base;
  const bf16_t* kp = k + base;
  const bf16_t* vp = vt + base;              // transposed layout [h][T]

  // Q A-tiles (loop invariant): two 16x32 chunks over h
  v16bf aq0, aq1;
  {
    const bf16_t* qrow = qp + (size_t)(qw + l15) * HS;
    #pragma unroll
    for (int i = 0; i < 8; ++i) {
      int kk = a_kbase(i, half4);
      aq0[2 * i]     = qrow[kk];
      aq0[2 * i + 1] = qrow[kk + 1];
      aq1[2 * i]     = qrow[32 + kk];
      aq1[2 * i + 1] = qrow[32 + kk + 1];
    }
  }

  v8f zero = {};
  v8f o[4];
  #pragma unroll
  for (int t = 0; t < 4; ++t) o[t] = zero;
  float rmax[8], rsum[8];
  #pragma unroll
  for (int i = 0; i < 8; ++i) { rmax[i] = -INFINITY; rsum[i] = 0.f; }

  const int nkb = (qw + 15) / 64 + 1;        // per-wave causal bound (64-key blocks)

  for (int kb = 0; kb < nkb; ++kb) {
    const int kbase = kb * 64;

    // ---- scores S = Q K^T : four 16x16 key sub-tiles ----
    v8f s[4];
    #pragma unroll
    for (int n = 0; n < 4; ++n) s[n] = zero;
    #pragma unroll
    for (int n = 0; n < 4; ++n) {
      const bf16_t* kr = kp + (size_t)(kbase + n * 16 + l15) * HS;
      v16bf b0, b1;
      #pragma unroll
      for (int j = 0; j < 16; ++j) {
        int kk = (j & 7) + 8 * half4 + ((j >> 3) << 4);
        b0[j] = kr[kk];
        b1[j] = kr[32 + kk];
      }
      s[n] = wmma_bf16(aq0, b0, s[n]);
      s[n] = wmma_bf16(aq1, b1, s[n]);
    }

    // ---- online softmax (per row: vgpr i, row = i + 8*half4) ----
    const float scale = 0.125f;              // 1/sqrt(64)
    float p[4][8], alpha[8];
    #pragma unroll
    for (int i = 0; i < 8; ++i) {
      int Q = qw + i + 8 * half4;
      float vv[4];
      float m = -INFINITY;
      #pragma unroll
      for (int n = 0; n < 4; ++n) {
        int key = kbase + n * 16 + l15;
        vv[n] = (key <= Q) ? s[n][i] * scale : -INFINITY;
        m = fmaxf(m, vv[n]);
      }
      #pragma unroll
      for (int d = 1; d < 16; d <<= 1) m = fmaxf(m, __shfl_xor(m, d, 32));
      float nm = fmaxf(rmax[i], m);
      float a  = __expf(rmax[i] - nm);
      float ss = 0.f;
      #pragma unroll
      for (int n = 0; n < 4; ++n) {
        float e = __expf(vv[n] - nm);
        p[n][i] = e;
        ss += e;
      }
      #pragma unroll
      for (int d = 1; d < 16; d <<= 1) ss += __shfl_xor(ss, d, 32);
      rsum[i]  = rsum[i] * a + ss;
      rmax[i]  = nm;
      alpha[i] = a;
    }

    // ---- C-layout -> A-layout transpose of P through LDS (same-wave DS in-order) --
    bf16_t* P = Pl[wave];
    #pragma unroll
    for (int i = 0; i < 8; ++i) {
      int row = i + 8 * half4;
      #pragma unroll
      for (int n = 0; n < 4; ++n)
        P[row * 64 + n * 16 + l15] = (bf16_t)p[n][i];
    }
    asm volatile("" ::: "memory");           // keep ds_read after ds_write
    v16bf ap0, ap1;                          // keys 0-31 / 32-63 of the block
    {
      const bf16_t* pr = P + l15 * 64;
      #pragma unroll
      for (int j = 0; j < 16; ++j) {
        ap0[j] = pr[16 * half4 + j];
        ap1[j] = pr[32 + 16 * half4 + j];
      }
    }

    // ---- O = O*alpha + P V  (V read directly from transposed global) ----
    #pragma unroll
    for (int t = 0; t < 4; ++t) {
      v8f ot = o[t];
      #pragma unroll
      for (int i = 0; i < 8; ++i) ot[i] *= alpha[i];
      const bf16_t* vrow = vp + (size_t)(t * 16 + l15) * TSEQ + kbase;
      v16bf bv0 = *(const v16bf*)(vrow + 16 * half4);       // keys 0-31
      v16bf bv1 = *(const v16bf*)(vrow + 32 + 16 * half4);  // keys 32-63
      ot   = wmma_bf16(ap0, bv0, ot);
      o[t] = wmma_bf16(ap1, bv1, ot);
    }
  }

  // ---- epilogue: normalize and store fp32 ----
  float* op = out + base;
  #pragma unroll
  for (int t = 0; t < 4; ++t)
    #pragma unroll
    for (int i = 0; i < 8; ++i) {
      int row = qw + i + 8 * half4;
      op[(size_t)row * HS + t * 16 + l15] = o[t][i] / rsum[i];
    }
}

extern "C" void kernel_launch(void* const* d_in, const int* in_sizes, int n_in,
                              void* d_out, int out_size, void* d_ws, size_t ws_size,
                              hipStream_t stream) {
  const float* x  = (const float*)d_in[0];
  const float* Wq = (const float*)d_in[1];
  const float* Wk = (const float*)d_in[2];
  const float* Wv = (const float*)d_in[3];
  float* out = (float*)d_out;

  const size_t nelem = (size_t)BATCH * TSEQ * HS;         // 2M elements
  bf16_t* qws = (bf16_t*)d_ws;
  bf16_t* kws = (bf16_t*)((char*)d_ws + nelem * sizeof(bf16_t));
  bf16_t* vws = (bf16_t*)((char*)d_ws + 2 * nelem * sizeof(bf16_t));  // [B][64][T]

  // 32768 rows / 128 rows-per-WG = 256 workgroups, 8 waves each
  proj_kernel<<<256, 256, 0, stream>>>(x, Wq, Wk, Wv, qws, kws, vws);
  // 8 batches x 64 query tiles = 512 workgroups, 4 waves each, barrier-free
  attn_kernel<<<512, 128, 0, stream>>>(qws, kws, vws, out);
}
